// GATNet_35296041238624
// MI455X (gfx1250) — matrix operand order repaired and checked by reading, not compiled, not run on hardware
//
#include <hip/hip_runtime.h>

#define N_NODES 50000
#define N_EDGES 800000
#define HF 128      // layer-1 width (H*F_H)
#define FOUT 64     // layer-2 width

typedef __attribute__((ext_vector_type(16))) __bf16 v16bf;
typedef __attribute__((ext_vector_type(8)))  float  v8f;

__device__ __forceinline__ unsigned short f2bf(float f) {
  unsigned int u = __float_as_uint(f);
  u += 0x7fffu + ((u >> 16) & 1u);           // round-to-nearest-even
  return (unsigned short)(u >> 16);
}
__device__ __forceinline__ float lrelu(float x) { return x > 0.f ? x : 0.2f * x; }

__device__ __forceinline__ void atomicMaxF(float* addr, float val) {
  if (val >= 0.f) atomicMax((int*)addr, __float_as_int(val));
  else            atomicMin((unsigned int*)addr, __float_as_uint(val));
}

// ---------------- prep kernels ----------------

__global__ void fill_f32(float* p, float v, int n) {
  int i = blockIdx.x * blockDim.x + threadIdx.x;
  if (i < n) p[i] = v;
}

__global__ void cvt_bf16(const float* __restrict__ in, unsigned short* __restrict__ out, int n) {
  int i = blockIdx.x * blockDim.x + threadIdx.x;
  if (i < n) out[i] = f2bf(in[i]);
}

// Pack W[K][NC] (row-major f32) into WMMA B-operand layout:
// P[((kt*NNT+nt)*32 + lane)*16 + e] = bf16( W[kt*32 + (e&7) + 8*(lane>=16) + 16*(e>>3)][nt*16 + (lane&15)] )
__global__ void pack_w(const float* __restrict__ W, unsigned short* __restrict__ P,
                       int NC, int total) {
  int idx = blockIdx.x * blockDim.x + threadIdx.x;
  if (idx >= total) return;
  int e    = idx & 15;
  int lane = (idx >> 4) & 31;
  int t    = idx >> 9;
  int NNT  = NC >> 4;
  int nt   = t % NNT;
  int kt   = t / NNT;
  int k = kt * 32 + (e & 7) + 8 * (lane >> 4) + 16 * (e >> 3);
  int n = nt * 16 + (lane & 15);
  P[idx] = f2bf(W[k * NC + n]);
}

// ---------------- WMMA GEMM ----------------
// One wave per 16-row stripe; computes X@Wl and X@Wr (shared, register-hoisted A).
// Loop order: nt outer, kt inner -> only 2 live accumulators (no spills).
// Xb: bf16 row-major [nTiles*16][128]. Bl/Br: packed B layout. K fixed = 128.

union ABu { uint4 u[2]; v16bf v; };

template<int NNT>
__global__ void gemm_dual(const unsigned short* __restrict__ Xb,
                          const unsigned short* __restrict__ Bl,
                          const unsigned short* __restrict__ Br,
                          float* __restrict__ XL, float* __restrict__ XR,
                          int nTiles) {
  int wid  = (int)((blockIdx.x * blockDim.x + threadIdx.x) >> 5);
  int lane = threadIdx.x & 31;
  if (wid >= nTiles) return;                 // wave-uniform: EXEC stays all-ones
  const int NC = NNT * 16;
  int m0     = wid * 16;
  int laneHi = lane >> 4;
  int m      = lane & 15;

  // Hoist all 4 A k-tiles (16x32 bf16 each): lane holds K = {k0+8*hi..+7} and {k0+16+8*hi..+7}
  ABu a[4];
  const unsigned short* abase = Xb + (size_t)(m0 + m) * 128 + 8 * laneHi;
#pragma unroll
  for (int kt = 0; kt < 4; ++kt) {
    a[kt].u[0] = *(const uint4*)(abase + kt * 32);
    a[kt].u[1] = *(const uint4*)(abase + kt * 32 + 16);
  }

#pragma unroll
  for (int nt = 0; nt < NNT; ++nt) {
    v8f accL = {};
    v8f accR = {};
#pragma unroll
    for (int kt = 0; kt < 4; ++kt) {
      const size_t boff = ((size_t)(kt * NNT + nt) * 32 + lane) * 16;
      ABu bl, br;
      const unsigned short* blp = Bl + boff;
      const unsigned short* brp = Br + boff;
      bl.u[0] = *(const uint4*)(blp);
      bl.u[1] = *(const uint4*)(blp + 8);
      br.u[0] = *(const uint4*)(brp);
      br.u[1] = *(const uint4*)(brp + 8);
      accL = __builtin_amdgcn_wmma_f32_16x16x32_bf16(
          false, a[kt].v, false, bl.v, (short)0, accL, false, false);
      accR = __builtin_amdgcn_wmma_f32_16x16x32_bf16(
          false, a[kt].v, false, br.v, (short)0, accR, false, false);
    }
    // D layout: VGPR r, lane L -> D[r + 8*(L>=16)][L&15]
    int col = nt * 16 + (lane & 15);
#pragma unroll
    for (int r = 0; r < 8; ++r) {
      int row = m0 + r + 8 * laneHi;
      XL[(size_t)row * NC + col] = accL[r];
      XR[(size_t)row * NC + col] = accR[r];
    }
  }
}

// ---------------- layer-1 edge sweeps (H=8 heads x 16 ch) ----------------

// a[e][h] = sum_c lrelu(XL[src][h,c]+XR[dst][h,c]) * att[h][c]; segment-max into m1.
__global__ void edge_att1(const int* __restrict__ ei,
                          const float* __restrict__ XL, const float* __restrict__ XR,
                          const float* __restrict__ att,
                          float* __restrict__ a1, float* __restrict__ m1) {
  int e    = (int)((blockIdx.x * blockDim.x + threadIdx.x) >> 5);
  int lane = threadIdx.x & 31;
  if (e >= N_EDGES) return;
  int src = ei[e];
  int dst = ei[N_EDGES + e];
  const float4 xl = *(const float4*)(XL + (size_t)src * HF + lane * 4);
  const float4 xr = *(const float4*)(XR + (size_t)dst * HF + lane * 4);
  const float4 at = *(const float4*)(att + lane * 4);
  float p = lrelu(xl.x + xr.x) * at.x + lrelu(xl.y + xr.y) * at.y +
            lrelu(xl.z + xr.z) * at.z + lrelu(xl.w + xr.w) * at.w;
  p += __shfl_xor(p, 1, 32);
  p += __shfl_xor(p, 2, 32);                 // reduce within 4-lane head group
  int h = lane >> 2;
  if ((lane & 3) == 0) {
    a1[(size_t)e * 8 + h] = p;
    atomicMaxF(m1 + (size_t)dst * 8 + h, p);
  }
}

__global__ void edge_softmax1(const int* __restrict__ ei, float* __restrict__ a1,
                              const float* __restrict__ m1, float* __restrict__ s1) {
  int i = blockIdx.x * blockDim.x + threadIdx.x;   // over E*8
  if (i >= N_EDGES * 8) return;
  int e = i >> 3, h = i & 7;
  int dst = ei[N_EDGES + e];
  float ea = expf(a1[i] - m1[(size_t)dst * 8 + h]);
  a1[i] = ea;
  atomicAdd(s1 + (size_t)dst * 8 + h, ea);
}

__global__ void edge_aggr1(const int* __restrict__ ei, const float* __restrict__ XL,
                           const float* __restrict__ a1, const float* __restrict__ s1,
                           float* __restrict__ o1) {
  int e    = (int)((blockIdx.x * blockDim.x + threadIdx.x) >> 5);
  int lane = threadIdx.x & 31;
  if (e >= N_EDGES) return;
  int src = ei[e];
  int dst = ei[N_EDGES + e];
  int h = lane >> 2;
  float coef = a1[(size_t)e * 8 + h] / (s1[(size_t)dst * 8 + h] + 1e-16f);
  const float4 xl = *(const float4*)(XL + (size_t)src * HF + lane * 4);
  float* op = o1 + (size_t)dst * HF + lane * 4;
  atomicAdd(op + 0, coef * xl.x);
  atomicAdd(op + 1, coef * xl.y);
  atomicAdd(op + 2, coef * xl.z);
  atomicAdd(op + 3, coef * xl.w);
}

// + bias, LayerNorm, ELU; emit bf16 h1 for the layer-2 GEMM. Wave per node.
__global__ void finalize1(const float* __restrict__ acc, const float* __restrict__ bias,
                          const float* __restrict__ g, const float* __restrict__ b,
                          unsigned short* __restrict__ h1b) {
  int node = (int)((blockIdx.x * blockDim.x + threadIdx.x) >> 5);
  int lane = threadIdx.x & 31;
  if (node >= N_NODES) return;
  const float4 v4 = *(const float4*)(acc + (size_t)node * HF + lane * 4);
  const float4 bi = *(const float4*)(bias + lane * 4);
  float v0 = v4.x + bi.x, v1 = v4.y + bi.y, v2 = v4.z + bi.z, v3 = v4.w + bi.w;
  float s = v0 + v1 + v2 + v3;
#pragma unroll
  for (int o = 16; o > 0; o >>= 1) s += __shfl_xor(s, o, 32);
  float mu = s * (1.0f / 128.0f);
  float d0 = v0 - mu, d1 = v1 - mu, d2 = v2 - mu, d3 = v3 - mu;
  float q = d0 * d0 + d1 * d1 + d2 * d2 + d3 * d3;
#pragma unroll
  for (int o = 16; o > 0; o >>= 1) q += __shfl_xor(q, o, 32);
  float rs = rsqrtf(q * (1.0f / 128.0f) + 1e-5f);
  const float4 gg = *(const float4*)(g + lane * 4);
  const float4 bb = *(const float4*)(b + lane * 4);
  float y0 = d0 * rs * gg.x + bb.x;
  float y1 = d1 * rs * gg.y + bb.y;
  float y2 = d2 * rs * gg.z + bb.z;
  float y3 = d3 * rs * gg.w + bb.w;
  y0 = y0 > 0.f ? y0 : expf(y0) - 1.f;
  y1 = y1 > 0.f ? y1 : expf(y1) - 1.f;
  y2 = y2 > 0.f ? y2 : expf(y2) - 1.f;
  y3 = y3 > 0.f ? y3 : expf(y3) - 1.f;
  uint2 pk;
  pk.x = (unsigned int)f2bf(y0) | ((unsigned int)f2bf(y1) << 16);
  pk.y = (unsigned int)f2bf(y2) | ((unsigned int)f2bf(y3) << 16);
  *(uint2*)(h1b + (size_t)node * HF + lane * 4) = pk;
}

// ---------------- layer-2 edge sweeps (1 head x 64 ch) ----------------

__global__ void edge_att2(const int* __restrict__ ei,
                          const float* __restrict__ XL, const float* __restrict__ XR,
                          const float* __restrict__ att,
                          float* __restrict__ a2, float* __restrict__ m2) {
  int e    = (int)((blockIdx.x * blockDim.x + threadIdx.x) >> 5);
  int lane = threadIdx.x & 31;
  if (e >= N_EDGES) return;
  int src = ei[e];
  int dst = ei[N_EDGES + e];
  const float2 xl = *(const float2*)(XL + (size_t)src * FOUT + lane * 2);
  const float2 xr = *(const float2*)(XR + (size_t)dst * FOUT + lane * 2);
  const float2 at = *(const float2*)(att + lane * 2);
  float p = lrelu(xl.x + xr.x) * at.x + lrelu(xl.y + xr.y) * at.y;
#pragma unroll
  for (int o = 16; o > 0; o >>= 1) p += __shfl_xor(p, o, 32);
  if (lane == 0) {
    a2[e] = p;
    atomicMaxF(m2 + dst, p);
  }
}

__global__ void edge_softmax2(const int* __restrict__ ei, float* __restrict__ a2,
                              const float* __restrict__ m2, float* __restrict__ s2) {
  int e = blockIdx.x * blockDim.x + threadIdx.x;
  if (e >= N_EDGES) return;
  int dst = ei[N_EDGES + e];
  float ea = expf(a2[e] - m2[dst]);
  a2[e] = ea;
  atomicAdd(s2 + dst, ea);
}

__global__ void edge_aggr2(const int* __restrict__ ei, const float* __restrict__ XL,
                           const float* __restrict__ a2, const float* __restrict__ s2,
                           float* __restrict__ o2) {
  int e    = (int)((blockIdx.x * blockDim.x + threadIdx.x) >> 5);
  int lane = threadIdx.x & 31;
  if (e >= N_EDGES) return;
  int src = ei[e];
  int dst = ei[N_EDGES + e];
  float coef = a2[e] / (s2[dst] + 1e-16f);
  const float2 xl = *(const float2*)(XL + (size_t)src * FOUT + lane * 2);
  float* op = o2 + (size_t)dst * FOUT + lane * 2;
  atomicAdd(op + 0, coef * xl.x);
  atomicAdd(op + 1, coef * xl.y);
}

__global__ void finalize2(const float* __restrict__ acc, const float* __restrict__ bias,
                          const float* __restrict__ g, const float* __restrict__ b,
                          float* __restrict__ out) {
  int node = (int)((blockIdx.x * blockDim.x + threadIdx.x) >> 5);
  int lane = threadIdx.x & 31;
  if (node >= N_NODES) return;
  const float2 v2 = *(const float2*)(acc + (size_t)node * FOUT + lane * 2);
  const float2 bi = *(const float2*)(bias + lane * 2);
  float v0 = v2.x + bi.x, v1 = v2.y + bi.y;
  float s = v0 + v1;
#pragma unroll
  for (int o = 16; o > 0; o >>= 1) s += __shfl_xor(s, o, 32);
  float mu = s * (1.0f / 64.0f);
  float d0 = v0 - mu, d1 = v1 - mu;
  float q = d0 * d0 + d1 * d1;
#pragma unroll
  for (int o = 16; o > 0; o >>= 1) q += __shfl_xor(q, o, 32);
  float rs = rsqrtf(q * (1.0f / 64.0f) + 1e-5f);
  const float2 gg = *(const float2*)(g + lane * 2);
  const float2 bb = *(const float2*)(b + lane * 2);
  float2 y;
  y.x = d0 * rs * gg.x + bb.x;
  y.y = d1 * rs * gg.y + bb.y;
  *(float2*)(out + (size_t)node * FOUT + lane * 2) = y;
}

// ---------------- launch ----------------

extern "C" void kernel_launch(void* const* d_in, const int* in_sizes, int n_in,
                              void* d_out, int out_size, void* d_ws, size_t ws_size,
                              hipStream_t stream) {
  const float* x     = (const float*)d_in[0];
  const int*   ei    = (const int*)d_in[1];
  const float* Wl1   = (const float*)d_in[2];
  const float* Wr1   = (const float*)d_in[3];
  const float* att1  = (const float*)d_in[4];
  const float* bias1 = (const float*)d_in[5];
  const float* g1    = (const float*)d_in[6];
  const float* b1    = (const float*)d_in[7];
  const float* Wl2   = (const float*)d_in[8];
  const float* Wr2   = (const float*)d_in[9];
  const float* att2  = (const float*)d_in[10];
  const float* bias2 = (const float*)d_in[11];
  const float* g2    = (const float*)d_in[12];
  const float* b2    = (const float*)d_in[13];
  float* out = (float*)d_out;
  (void)in_sizes; (void)n_in; (void)out_size; (void)ws_size;

  char* base = (char*)d_ws;
  size_t off = 0;
  auto take = [&](size_t bytes) -> void* {
    void* p = base + off;
    off += (bytes + 255) & ~(size_t)255;
    return p;
  };
  unsigned short* xb   = (unsigned short*)take((size_t)N_NODES * 128 * 2);
  unsigned short* wl1p = (unsigned short*)take((size_t)128 * 128 * 2);
  unsigned short* wr1p = (unsigned short*)take((size_t)128 * 128 * 2);
  unsigned short* wl2p = (unsigned short*)take((size_t)128 * 64 * 2);
  unsigned short* wr2p = (unsigned short*)take((size_t)128 * 64 * 2);
  unsigned short* h1b  = (unsigned short*)take((size_t)N_NODES * 128 * 2);
  float* XL1 = (float*)take((size_t)N_NODES * 128 * 4);
  float* XR1 = (float*)take((size_t)N_NODES * 128 * 4);
  float* a1  = (float*)take((size_t)N_EDGES * 8 * 4);
  float* m1  = (float*)take((size_t)N_NODES * 8 * 4);
  float* s1  = (float*)take((size_t)N_NODES * 8 * 4);
  float* o1  = (float*)take((size_t)N_NODES * 128 * 4);
  float* XL2 = (float*)take((size_t)N_NODES * 64 * 4);
  float* XR2 = (float*)take((size_t)N_NODES * 64 * 4);
  float* a2  = (float*)take((size_t)N_EDGES * 4);
  float* m2  = (float*)take((size_t)N_NODES * 4);
  float* s2  = (float*)take((size_t)N_NODES * 4);
  float* o2  = (float*)take((size_t)N_NODES * 64 * 4);

  // prep: bf16 conversion + WMMA-layout weight packing + accumulator init
  cvt_bf16<<<(N_NODES * 128 + 255) / 256, 256, 0, stream>>>(x, xb, N_NODES * 128);
  pack_w<<<(128 * 128 + 255) / 256, 256, 0, stream>>>(Wl1, wl1p, 128, 128 * 128);
  pack_w<<<(128 * 128 + 255) / 256, 256, 0, stream>>>(Wr1, wr1p, 128, 128 * 128);
  pack_w<<<(128 * 64 + 255) / 256, 256, 0, stream>>>(Wl2, wl2p, 64, 128 * 64);
  pack_w<<<(128 * 64 + 255) / 256, 256, 0, stream>>>(Wr2, wr2p, 64, 128 * 64);
  fill_f32<<<(N_NODES * 8 + 255) / 256, 256, 0, stream>>>(m1, -3.0e38f, N_NODES * 8);
  fill_f32<<<(N_NODES * 8 + 255) / 256, 256, 0, stream>>>(s1, 0.f, N_NODES * 8);
  fill_f32<<<(N_NODES * 128 + 255) / 256, 256, 0, stream>>>(o1, 0.f, N_NODES * 128);
  fill_f32<<<(N_NODES + 255) / 256, 256, 0, stream>>>(m2, -3.0e38f, N_NODES);
  fill_f32<<<(N_NODES + 255) / 256, 256, 0, stream>>>(s2, 0.f, N_NODES);
  fill_f32<<<(N_NODES * 64 + 255) / 256, 256, 0, stream>>>(o2, 0.f, N_NODES * 64);

  const int nTiles = N_NODES / 16;  // 3125 (exact)

  // layer 1
  gemm_dual<8><<<(nTiles + 3) / 4, 128, 0, stream>>>(xb, wl1p, wr1p, XL1, XR1, nTiles);
  edge_att1<<<N_EDGES / 8, 256, 0, stream>>>(ei, XL1, XR1, att1, a1, m1);
  edge_softmax1<<<(N_EDGES * 8) / 256, 256, 0, stream>>>(ei, a1, m1, s1);
  edge_aggr1<<<N_EDGES / 8, 256, 0, stream>>>(ei, XL1, a1, s1, o1);
  finalize1<<<(N_NODES * 32) / 256, 256, 0, stream>>>(o1, bias1, g1, b1, h1b);

  // layer 2
  gemm_dual<4><<<(nTiles + 3) / 4, 128, 0, stream>>>(h1b, wl2p, wr2p, XL2, XR2, nTiles);
  edge_att2<<<N_EDGES / 8, 256, 0, stream>>>(ei, XL2, XR2, att2, a2, m2);
  edge_softmax2<<<(N_EDGES + 255) / 256, 256, 0, stream>>>(ei, a2, m2, s2);
  edge_aggr2<<<N_EDGES / 8, 256, 0, stream>>>(ei, XL2, a2, s2, o2);
  finalize2<<<(N_NODES * 32) / 256, 256, 0, stream>>>(o2, bias2, g2, b2, out);
}
